// dec_MultiHeadedAttention_46059229282740
// MI455X (gfx1250) — matrix operand order repaired and checked by reading, not compiled, not run on hardware
//
#include <hip/hip_runtime.h>
#include <hip/hip_bf16.h>

typedef __attribute__((ext_vector_type(16))) _Float16 v16h;
typedef __attribute__((ext_vector_type(8)))  _Float16 v8h;
typedef __attribute__((ext_vector_type(8)))  float    v8f;

#define H_    12
#define DM    768
#define DK    64
#define B_    2
#define S_    2048
#define M_TOT (B_ * S_)   // 4096

union ABf { v16h v; v8h h[2]; };

__device__ __forceinline__ v8f wmma_f16(const ABf& a, const ABf& b, v8f c) {
    // D = A(16x32 f16) * B(32x16 f16) + C(16x16 f32)
    return __builtin_amdgcn_wmma_f32_16x16x32_f16(
        /*neg_a=*/false, a.v, /*neg_b=*/false, b.v,
        /*c_mod=*/(short)0, c, /*reuse_a=*/false, /*reuse_b=*/false);
}

// ds_swizzle group-of-32: offset = xor_mask<<10 | or<<5 | and_mask(0x1f)
template<int OFF>
__device__ __forceinline__ float swz(float v) {
    return __int_as_float(__builtin_amdgcn_ds_swizzle(__float_as_int(v), OFF));
}

// ---------------- f32 -> f16 convert ----------------
__global__ void k_cvt(const float* __restrict__ in, _Float16* __restrict__ out, int n) {
    int i = blockIdx.x * blockDim.x + threadIdx.x;
    int stride = gridDim.x * blockDim.x;
    for (; i < n; i += stride) out[i] = (_Float16)in[i];
}

// ============ shared GEMM mainloop: C[32x64] per wave, W staged in LDS ============
// Block = 8 waves sharing one 64-column N range (256 M rows total).
// Weight chunk (64 N rows x 64 k, 8 KB f16) is copied global->LDS with
// GLOBAL_LOAD_ASYNC_TO_LDS_B128 (ASYNCcnt), double-buffered: copy of chunk c+1
// overlaps WMMAs of chunk c. B-fragments are read from LDS (ds_load_b128).
__device__ __forceinline__ void gemm_mainloop(const _Float16* __restrict__ X,
                                              const _Float16* __restrict__ Wm,
                                              _Float16* ldsW,          // 2 x 4096 halves
                                              int m0, int n0, int tidx,
                                              v8f acc0[4], v8f acc1[4]) {
    const int lane = tidx & 31, lm = lane & 15, hi = lane >> 4;
    const _Float16* arow0 = X + (size_t)(m0 + lm) * DM + hi * 8;
    const _Float16* arow1 = arow0 + (size_t)16 * DM;
    const unsigned ldsBase = (unsigned)(uintptr_t)ldsW;  // group-segment offset

    auto issue_chunk = [&](int buf, int kc) {
        // 8 KB chunk = 512 x 16B; 256 threads x 2 async b128 copies
#pragma unroll
        for (int i = 0; i < 2; ++i) {
            const int id  = tidx * 2 + i;
            const int row = id >> 3;   // 0..63 : W row (n0+row)
            const int seg = id & 7;    // 16-byte segment within the 128B row
            const unsigned long long g = (unsigned long long)(uintptr_t)
                ((const char*)(Wm + (size_t)(n0 + row) * DM + kc) + seg * 16);
            const unsigned l = ldsBase + (unsigned)(buf * 8192 + row * 128 + seg * 16);
            asm volatile("global_load_async_to_lds_b128 %0, %1, off"
                         :: "v"(l), "v"(g) : "memory");
        }
    };

    issue_chunk(0, 0);
    for (int c = 0; c < DM / 64; ++c) {
        asm volatile("s_wait_asynccnt 0x0" ::: "memory"); // our copies of chunk c landed
        __syncthreads();                                  // everyone's copies landed;
                                                          // buffer (c+1)&1 is free
        if (c + 1 < DM / 64) issue_chunk((c + 1) & 1, (c + 1) * 64);
        const _Float16* bufp = ldsW + (c & 1) * 4096;
#pragma unroll
        for (int ks = 0; ks < 64; ks += 32) {
            ABf a0, a1;
            a0.h[0] = *(const v8h*)(arow0 + c * 64 + ks);
            a0.h[1] = *(const v8h*)(arow0 + c * 64 + ks + 16);
            a1.h[0] = *(const v8h*)(arow1 + c * 64 + ks);
            a1.h[1] = *(const v8h*)(arow1 + c * 64 + ks + 16);
#pragma unroll
            for (int nf = 0; nf < 4; ++nf) {
                const _Float16* bp = bufp + (nf * 16 + lm) * 64 + ks + hi * 8;
                ABf b;
                b.h[0] = *(const v8h*)(bp);       // ds_load_b128
                b.h[1] = *(const v8h*)(bp + 16);
                acc0[nf] = wmma_f16(a0, b, acc0[nf]);
                acc1[nf] = wmma_f16(a1, b, acc1[nf]);
            }
        }
    }
}

// ---------------- fused QKV projection: X @ W^T + b ----------------
// blockIdx.z selects Q(0)/K(1)/V(2).
// Q,K stored [b][h][s][d]; V stored transposed [b][h][d][s] for PV WMMA B-frags.
__global__ void k_gemm_qkv(const _Float16* __restrict__ xq, const _Float16* __restrict__ xk,
                           const _Float16* __restrict__ xv,
                           const _Float16* __restrict__ wq, const _Float16* __restrict__ wk,
                           const _Float16* __restrict__ wv,
                           const float* __restrict__ bq, const float* __restrict__ bk,
                           const float* __restrict__ bv,
                           _Float16* __restrict__ q16, _Float16* __restrict__ k16,
                           _Float16* __restrict__ vt16) {
    __shared__ __align__(16) _Float16 ldsW[2 * 4096];   // 16 KB double buffer
    const int z = blockIdx.z;
    const _Float16* X    = (z == 0) ? xq : (z == 1) ? xk : xv;
    const _Float16* W    = (z == 0) ? wq : (z == 1) ? wk : wv;
    const float*    bias = (z == 0) ? bq : (z == 1) ? bk : bv;

    const int wid = threadIdx.x >> 5, lane = threadIdx.x & 31;
    const int lm = lane & 15, hi = lane >> 4;
    const int nT = blockIdx.x % (DM / 64);
    const int mB = blockIdx.x / (DM / 64);
    const int m0 = mB * 256 + wid * 32;
    const int n0 = nT * 64;

    v8f acc0[4] = {}, acc1[4] = {};
    gemm_mainloop(X, W, ldsW, m0, n0, threadIdx.x, acc0, acc1);

    const int bIdx = m0 / S_;                 // 256 | 2048, no straddle
    const int s0   = m0 % S_;
#pragma unroll
    for (int nf = 0; nf < 4; ++nf) {
        const int n = n0 + nf * 16 + lm;
        const float bb = bias[n];
        const int h = n / DK, d = n % DK;
        if (z == 2) {
            v8h pk0, pk1;
#pragma unroll
            for (int r = 0; r < 8; ++r) {
                pk0[r] = (_Float16)(acc0[nf][r] + bb);
                pk1[r] = (_Float16)(acc1[nf][r] + bb);
            }
            _Float16* vdst = vt16 + ((size_t)(bIdx * H_ + h) * DK + d) * S_;
            *(v8h*)(vdst + s0 + hi * 8)      = pk0;   // rows are consecutive s
            *(v8h*)(vdst + s0 + 16 + hi * 8) = pk1;
        } else {
            _Float16* dst = ((z == 0) ? q16 : k16) + (size_t)(bIdx * H_ + h) * S_ * DK + d;
#pragma unroll
            for (int r = 0; r < 8; ++r) {
                dst[(size_t)(s0 + r + hi * 8) * DK]      = (_Float16)(acc0[nf][r] + bb);
                dst[(size_t)(s0 + 16 + r + hi * 8) * DK] = (_Float16)(acc1[nf][r] + bb);
            }
        }
    }
}

// ---------------- flash attention with causal mask + time-decay bias ----------------
__global__ void k_attn(const _Float16* __restrict__ q16, const _Float16* __restrict__ k16,
                       const _Float16* __restrict__ vt16, _Float16* __restrict__ o16) {
    __shared__ __align__(16) _Float16 ldsP[8][16 * 32];  // per-wave P tile (1 KB each)
    const int wid = threadIdx.x >> 5, lane = threadIdx.x & 31;
    const int lm = lane & 15, hi = lane >> 4;
    const int tid = blockIdx.x * 8 + wid;       // 3072 wave tiles
    const int qt = tid % (S_ / 16);
    const int bh = tid / (S_ / 16);
    const int q0 = qt * 16;

    const _Float16* qb = q16 + (size_t)bh * S_ * DK;
    const _Float16* kb = k16 + (size_t)bh * S_ * DK;
    const _Float16* vb = vt16 + (size_t)bh * DK * S_;   // [d][s]

    // Q A-fragments for K-dim 0..31 and 32..63
    const _Float16* qp = qb + (size_t)(q0 + lm) * DK + hi * 8;
    ABf aq0, aq1;
    aq0.h[0] = *(const v8h*)(qp);      aq0.h[1] = *(const v8h*)(qp + 16);
    aq1.h[0] = *(const v8h*)(qp + 32); aq1.h[1] = *(const v8h*)(qp + 48);

    v8f o[4] = {};
    float rmax[8], rsum[8];
#pragma unroll
    for (int r = 0; r < 8; ++r) { rmax[r] = -1e30f; rsum[r] = 0.f; }

    const float scale = 0.125f;                 // 1/sqrt(64)
    const int nkb = ((q0 + 15) >> 5) + 1;       // causal: only needed key blocks
    _Float16* pl = &ldsP[wid][0];

    for (int kbI = 0; kbI < nkb; ++kbI) {
        const int kk = kbI * 32;
        v8f sfr[2];
#pragma unroll
        for (int f = 0; f < 2; ++f) {           // scores = Q K^T over 32 keys
            const int key = kk + f * 16 + lm;
            const _Float16* bp = kb + (size_t)key * DK + hi * 8;
            ABf b0, b1;
            b0.h[0] = *(const v8h*)(bp);      b0.h[1] = *(const v8h*)(bp + 16);
            b1.h[0] = *(const v8h*)(bp + 32); b1.h[1] = *(const v8h*)(bp + 48);
            if (kbI + 1 < nkb)                 // prefetch next key block of K
                __builtin_prefetch(bp + (size_t)32 * DK, 0, 1);
            v8f c = {};
            c = wmma_f16(aq0, b0, c);
            c = wmma_f16(aq1, b1, c);
            sfr[f] = c;
        }
        // scale, causal mask, time-decay bias
#pragma unroll
        for (int f = 0; f < 2; ++f) {
            const int j = kk + f * 16 + lm;
            const float tb = __expf(-(float)(j + 1));
#pragma unroll
            for (int r = 0; r < 8; ++r) {
                const int mq = q0 + r + hi * 8;
                float s = sfr[f][r] * scale;
                s = (j > mq) ? -1e9f : s;
                sfr[f][r] = s - tb;
            }
        }
        // online-softmax row max (reduce across 16-lane N group via ds_swizzle xor)
        float nmax[8], fac[8];
#pragma unroll
        for (int r = 0; r < 8; ++r) {
            float m = fmaxf(sfr[0][r], sfr[1][r]);
            m = fmaxf(m, swz<0x041F>(m));
            m = fmaxf(m, swz<0x081F>(m));
            m = fmaxf(m, swz<0x101F>(m));
            m = fmaxf(m, swz<0x201F>(m));
            nmax[r] = fmaxf(rmax[r], m);
            fac[r]  = __expf(rmax[r] - nmax[r]);
            rmax[r] = nmax[r];
        }
        // P = exp(s - max), write f16 P tile to per-wave LDS (C-layout -> memory)
#pragma unroll
        for (int f = 0; f < 2; ++f) {
#pragma unroll
            for (int r = 0; r < 8; ++r) {
                const float p = __expf(sfr[f][r] - nmax[r]);
                sfr[f][r] = p;
                pl[(r + hi * 8) * 32 + f * 16 + lm] = (_Float16)p;
            }
        }
        // row sums + rescale running sum / accumulators
#pragma unroll
        for (int r = 0; r < 8; ++r) {
            float s = sfr[0][r] + sfr[1][r];
            s += swz<0x041F>(s);
            s += swz<0x081F>(s);
            s += swz<0x101F>(s);
            s += swz<0x201F>(s);
            rsum[r] = rsum[r] * fac[r] + s;
        }
#pragma unroll
        for (int nf = 0; nf < 4; ++nf)
#pragma unroll
            for (int r = 0; r < 8; ++r) o[nf][r] *= fac[r];

        asm volatile("s_wait_dscnt 0" ::: "memory");
        // reload P as A-fragment (A-layout) from LDS
        ABf ap;
        const _Float16* lp = pl + lm * 32 + hi * 8;
        ap.h[0] = *(const v8h*)(lp);
        ap.h[1] = *(const v8h*)(lp + 16);
        // O += P @ V  (B-frags contiguous thanks to V^T layout)
#pragma unroll
        for (int nf = 0; nf < 4; ++nf) {
            const _Float16* vp = vb + (size_t)(nf * 16 + lm) * S_ + kk + hi * 8;
            ABf b;
            b.h[0] = *(const v8h*)(vp);
            b.h[1] = *(const v8h*)(vp + 16);
            if (kbI + 1 < nkb)                 // prefetch next key block of V^T
                __builtin_prefetch(vp + 32, 0, 1);
            o[nf] = wmma_f16(ap, b, o[nf]);
        }
    }

    // normalize + store O in [b][s][h][d] (row-major 4096x768 for final GEMM)
    const int b = bh / H_, h = bh % H_;
#pragma unroll
    for (int nf = 0; nf < 4; ++nf) {
        const int d = nf * 16 + lm;
#pragma unroll
        for (int r = 0; r < 8; ++r) {
            const int s = q0 + r + hi * 8;
            o16[(((size_t)b * S_ + s) * H_ + h) * DK + d] = (_Float16)(o[nf][r] / rsum[r]);
        }
    }
}

// ---------------- output projection: O @ Wo^T + bo -> f32 ----------------
__global__ void k_gemm_out(const _Float16* __restrict__ o16, const _Float16* __restrict__ wo,
                           const float* __restrict__ bo, float* __restrict__ out) {
    __shared__ __align__(16) _Float16 ldsW[2 * 4096];   // 16 KB double buffer
    const int wid = threadIdx.x >> 5, lane = threadIdx.x & 31;
    const int lm = lane & 15, hi = lane >> 4;
    const int nT = blockIdx.x % (DM / 64);
    const int mB = blockIdx.x / (DM / 64);
    const int m0 = mB * 256 + wid * 32;
    const int n0 = nT * 64;

    v8f acc0[4] = {}, acc1[4] = {};
    gemm_mainloop(o16, wo, ldsW, m0, n0, threadIdx.x, acc0, acc1);

#pragma unroll
    for (int nf = 0; nf < 4; ++nf) {
        const int n = n0 + nf * 16 + lm;
        const float bb = bo[n];
#pragma unroll
        for (int r = 0; r < 8; ++r) {
            out[(size_t)(m0 + r + hi * 8) * DM + n]      = acc0[nf][r] + bb;
            out[(size_t)(m0 + 16 + r + hi * 8) * DM + n] = acc1[nf][r] + bb;
        }
    }
}

extern "C" void kernel_launch(void* const* d_in, const int* in_sizes, int n_in,
                              void* d_out, int out_size, void* d_ws, size_t ws_size,
                              hipStream_t stream) {
    (void)in_sizes; (void)n_in; (void)out_size; (void)ws_size;
    const float* query = (const float*)d_in[0];
    const float* key   = (const float*)d_in[1];
    const float* value = (const float*)d_in[2];
    // d_in[3] = mask: causal by construction, applied analytically in k_attn
    const float* Wq = (const float*)d_in[4];
    const float* bq = (const float*)d_in[5];
    const float* Wk = (const float*)d_in[6];
    const float* bk = (const float*)d_in[7];
    const float* Wv = (const float*)d_in[8];
    const float* bv = (const float*)d_in[9];
    const float* Wo = (const float*)d_in[10];
    const float* bo = (const float*)d_in[11];

    const size_t NX = (size_t)M_TOT * DM;   // 3,145,728
    const size_t NW = (size_t)DM * DM;      //   589,824
    _Float16* ws   = (_Float16*)d_ws;       // total ~48.8 MB of workspace
    _Float16* xq16 = ws;
    _Float16* xk16 = xq16 + NX;
    _Float16* xv16 = xk16 + NX;
    _Float16* wq16 = xv16 + NX;
    _Float16* wk16 = wq16 + NW;
    _Float16* wv16 = wk16 + NW;
    _Float16* wo16 = wv16 + NW;
    _Float16* q16  = wo16 + NW;
    _Float16* k16  = q16 + NX;
    _Float16* vt16 = k16 + NX;
    _Float16* o16  = vt16 + NX;

    k_cvt<<<2048, 256, 0, stream>>>(query, xq16, (int)NX);
    k_cvt<<<2048, 256, 0, stream>>>(key,   xk16, (int)NX);
    k_cvt<<<2048, 256, 0, stream>>>(value, xv16, (int)NX);
    k_cvt<<<512,  256, 0, stream>>>(Wq, wq16, (int)NW);
    k_cvt<<<512,  256, 0, stream>>>(Wk, wk16, (int)NW);
    k_cvt<<<512,  256, 0, stream>>>(Wv, wv16, (int)NW);
    k_cvt<<<512,  256, 0, stream>>>(Wo, wo16, (int)NW);

    dim3 gq(192, 1, 3);   // 16 Mx12 N blocks, 8 waves/block (32 rows per wave)
    k_gemm_qkv<<<gq, 256, 0, stream>>>(xq16, xk16, xv16, wq16, wk16, wv16,
                                       bq, bk, bv, q16, k16, vt16);
    k_attn<<<384, 256, 0, stream>>>(q16, k16, vt16, o16);
    k_gemm_out<<<192, 256, 0, stream>>>(o16, wo16, bo, (float*)d_out);
}
